// CausalSelfAttention_2121713844448
// MI455X (gfx1250) — matrix operand order repaired and checked by reading, not compiled, not run on hardware
//
#include <hip/hip_runtime.h>

// Problem dims (match reference)
#define B_ 8
#define T_ 2048
#define D_ 1024   // IN_DIM
#define E_ 1024   // EMB

typedef __attribute__((ext_vector_type(16))) __bf16 v16bf;
typedef __attribute__((ext_vector_type(8)))  float  v8f;

union BFrag {
  v16bf v;
  unsigned int u[8];
  uint4 q[2];
};

__device__ __forceinline__ unsigned short f2bf(float f) {
  unsigned int u = __float_as_uint(f);
  unsigned int r = u + 0x7FFFu + ((u >> 16) & 1u);  // RNE truncate to bf16
  return (unsigned short)(r >> 16);
}

// A operand (16x32 bf16, MxK). Lanes 0-15 hold K=0..15,
// lanes 16-31 hold K=16..31; row M = lane%16. Two contiguous 16B loads.
__device__ __forceinline__ void load_fragA(const unsigned short* __restrict__ base,
                                           int row0, int stride, int k0, int lane,
                                           BFrag& f) {
  const int hl = lane >> 4;
  const unsigned short* p = base + (size_t)(row0 + (lane & 15)) * stride + (k0 + 16 * hl);
  f.q[0] = *(const uint4*)p;         // K = 16*hl .. 16*hl+7
  f.q[1] = *(const uint4*)(p + 8);   // K = 16*hl+8 .. 16*hl+15
}

// B operand (32x16 bf16, KxN). Lanes 0-15: K=0..15; lanes 16-31: K=16..31;
// column N = lane%16. Source laid out [N][K] (transposed) -> one 32B run.
__device__ __forceinline__ void load_fragB(const unsigned short* __restrict__ base,
                                           int row0, int stride, int k0, int lane,
                                           BFrag& f) {
  const int hl = lane >> 4;
  const unsigned short* p = base + (size_t)(row0 + (lane & 15)) * stride + (k0 + 16 * hl);
  f.q[0] = *(const uint4*)p;
  f.q[1] = *(const uint4*)(p + 8);
}

__device__ __forceinline__ v8f wmma_bf16(const BFrag& a, const BFrag& b, v8f c) {
  return __builtin_amdgcn_wmma_f32_16x16x32_bf16(false, a.v, false, b.v,
                                                 (short)0, c, false, false);
}

// ---------------- init m/Z ----------------
__global__ void k_init(unsigned int* m_enc, float* Z) {
  int i = threadIdx.x;
  if (i < B_) { m_enc[i] = 0u; Z[i] = 0.0f; }
}

// ---------------- fp32 -> bf16 converts ----------------
__global__ void k_cvt_x(const float* __restrict__ x, unsigned short* __restrict__ Xb, int n4) {
  int i = blockIdx.x * blockDim.x + threadIdx.x;
  if (i >= n4) return;
  float4 v = ((const float4*)x)[i];
  ushort4 h;
  h.x = f2bf(v.x); h.y = f2bf(v.y); h.z = f2bf(v.z); h.w = f2bf(v.w);
  ((ushort4*)Xb)[i] = h;
}

__global__ void k_cvt_wt(const float* __restrict__ Wq, const float* __restrict__ Wk,
                         const float* __restrict__ Wv, unsigned short* __restrict__ Wt) {
  int idx = blockIdx.x * blockDim.x + threadIdx.x;   // 0 .. D_*E_-1
  int mat = blockIdx.y;
  const float* W = (mat == 0) ? Wq : (mat == 1) ? Wk : Wv;
  int k = idx >> 10;          // row of W
  int n = idx & 1023;         // col of W (coalesced read)
  Wt[(size_t)mat * D_ * E_ + (size_t)n * D_ + k] = f2bf(W[idx]);
}

// ---------------- QKV projection GEMM (128x128 block, 4 waves of 64x64) ----------------
__global__ __launch_bounds__(128) void k_qkv(const unsigned short* __restrict__ Xb,
                                             const unsigned short* __restrict__ Wt,
                                             const float* __restrict__ bq,
                                             const float* __restrict__ bk,
                                             const float* __restrict__ bv,
                                             unsigned short* __restrict__ Qb,
                                             unsigned short* __restrict__ Kb,
                                             unsigned short* __restrict__ Vt) {
  const int lane = threadIdx.x & 31;
  const int wave = threadIdx.x >> 5;
  const int wr = wave >> 1, wc = wave & 1;
  const int Mbase = blockIdx.y * 128 + wr * 64;
  const int Nbase = blockIdx.x * 128 + wc * 64;
  const int mat = blockIdx.z;
  const unsigned short* W = Wt + (size_t)mat * D_ * E_;

  v8f acc[4][4] = {};
  for (int k0 = 0; k0 < D_; k0 += 32) {
    BFrag a[4], b[4];
#pragma unroll
    for (int mr = 0; mr < 4; ++mr) load_fragA(Xb, Mbase + mr * 16, D_, k0, lane, a[mr]);
#pragma unroll
    for (int nc = 0; nc < 4; ++nc) load_fragB(W, Nbase + nc * 16, D_, k0, lane, b[nc]);
#pragma unroll
    for (int mr = 0; mr < 4; ++mr)
#pragma unroll
      for (int nc = 0; nc < 4; ++nc)
        acc[mr][nc] = wmma_bf16(a[mr], b[nc], acc[mr][nc]);
  }

  const float* bias = (mat == 0) ? bq : (mat == 1) ? bk : bv;
  const float qscale = 0.03125f;  // 1/sqrt(EMB)
#pragma unroll
  for (int mr = 0; mr < 4; ++mr)
#pragma unroll
    for (int nc = 0; nc < 4; ++nc)
#pragma unroll
      for (int r = 0; r < 8; ++r) {
        int M = Mbase + mr * 16 + r + 8 * (lane >> 4);
        int N = Nbase + nc * 16 + (lane & 15);
        float val = acc[mr][nc][r] + bias[N];
        if (mat == 0) {
          Qb[(size_t)M * E_ + N] = f2bf(val * qscale);
        } else if (mat == 1) {
          Kb[(size_t)M * E_ + N] = f2bf(val);
        } else {
          int bb = M >> 11, tl = M & (T_ - 1);
          Vt[((size_t)bb * E_ + N) * T_ + tl] = f2bf(val);  // store transposed
        }
      }
}

// ---------------- shared QK^T tile (one wave: 64x64) ----------------
__device__ __forceinline__ void qk_tile(const unsigned short* __restrict__ Qb,
                                        const unsigned short* __restrict__ Kb,
                                        int b, int tBase, int sBase, int lane,
                                        v8f acc[4][4]) {
  const unsigned short* Q = Qb + (size_t)b * T_ * D_;
  const unsigned short* K = Kb + (size_t)b * T_ * D_;
  for (int k0 = 0; k0 < D_; k0 += 32) {
    BFrag a[4], bb[4];
#pragma unroll
    for (int mr = 0; mr < 4; ++mr) load_fragA(Q, tBase + mr * 16, D_, k0, lane, a[mr]);
#pragma unroll
    for (int nc = 0; nc < 4; ++nc) load_fragB(K, sBase + nc * 16, D_, k0, lane, bb[nc]);
#pragma unroll
    for (int mr = 0; mr < 4; ++mr)
#pragma unroll
      for (int nc = 0; nc < 4; ++nc)
        acc[mr][nc] = wmma_bf16(a[mr], bb[nc], acc[mr][nc]);
  }
}

// ---------------- pass A: global masked max ----------------
__global__ __launch_bounds__(128) void k_smax(const unsigned short* __restrict__ Qb,
                                              const unsigned short* __restrict__ Kb,
                                              unsigned int* __restrict__ m_enc) {
  const int sT = blockIdx.x, tT = blockIdx.y, b = blockIdx.z;
  if (sT > tT) return;  // fully above diagonal
  const int lane = threadIdx.x & 31;
  const int wave = threadIdx.x >> 5;
  const int wr = wave >> 1, wc = wave & 1;
  const int tBase = tT * 128 + wr * 64;
  const int sBase = sT * 128 + wc * 64;

  v8f acc[4][4] = {};
  qk_tile(Qb, Kb, b, tBase, sBase, lane, acc);

  float mx = -3.0e38f;
#pragma unroll
  for (int mr = 0; mr < 4; ++mr)
#pragma unroll
    for (int nc = 0; nc < 4; ++nc)
#pragma unroll
      for (int r = 0; r < 8; ++r) {
        int t = tBase + mr * 16 + r + 8 * (lane >> 4);
        int s = sBase + nc * 16 + (lane & 15);
        if (s <= t) mx = fmaxf(mx, acc[mr][nc][r]);
      }
  for (int off = 16; off > 0; off >>= 1) mx = fmaxf(mx, __shfl_xor(mx, off, 32));
  __shared__ float red[4];
  if (lane == 0) red[wave] = mx;
  __syncthreads();
  if (threadIdx.x == 0) {
    float m2 = fmaxf(fmaxf(red[0], red[1]), fmaxf(red[2], red[3]));
    unsigned int u = __float_as_uint(m2);
    u = (u & 0x80000000u) ? ~u : (u | 0x80000000u);  // order-preserving encode
    atomicMax(&m_enc[b], u);
  }
}

// ---------------- pass B: P = exp(S - m), Z partial sums ----------------
__global__ __launch_bounds__(128) void k_pexp(const unsigned short* __restrict__ Qb,
                                              const unsigned short* __restrict__ Kb,
                                              const unsigned int* __restrict__ m_enc,
                                              unsigned short* __restrict__ P,
                                              float* __restrict__ Z) {
  const int sT = blockIdx.x, tT = blockIdx.y, b = blockIdx.z;
  if (sT > tT) return;
  const int lane = threadIdx.x & 31;
  const int wave = threadIdx.x >> 5;
  const int wr = wave >> 1, wc = wave & 1;
  const int tBase = tT * 128 + wr * 64;
  const int sBase = sT * 128 + wc * 64;

  v8f acc[4][4] = {};
  qk_tile(Qb, Kb, b, tBase, sBase, lane, acc);

  unsigned int eu = m_enc[b];
  float mglob = (eu & 0x80000000u) ? __uint_as_float(eu ^ 0x80000000u)
                                   : __uint_as_float(~eu);
  unsigned short* Pb = P + (size_t)b * T_ * T_;
  float zsum = 0.0f;
#pragma unroll
  for (int mr = 0; mr < 4; ++mr)
#pragma unroll
    for (int nc = 0; nc < 4; ++nc)
#pragma unroll
      for (int r = 0; r < 8; ++r) {
        int t = tBase + mr * 16 + r + 8 * (lane >> 4);
        int s = sBase + nc * 16 + (lane & 15);
        float p = 0.0f;
        if (s <= t) p = __expf(acc[mr][nc][r] - mglob);
        zsum += p;
        Pb[(size_t)t * T_ + s] = f2bf(p);
      }
  for (int off = 16; off > 0; off >>= 1) zsum += __shfl_xor(zsum, off, 32);
  __shared__ float red[4];
  if (lane == 0) red[wave] = zsum;
  __syncthreads();
  if (threadIdx.x == 0) {
    atomicAdd(&Z[b], red[0] + red[1] + red[2] + red[3]);
  }
}

// ---------------- O = (P @ V) / Z ----------------
__global__ __launch_bounds__(128) void k_pv(const unsigned short* __restrict__ P,
                                            const unsigned short* __restrict__ Vt,
                                            const float* __restrict__ Z,
                                            float* __restrict__ out) {
  const int eT = blockIdx.x, tT = blockIdx.y, b = blockIdx.z;
  const int lane = threadIdx.x & 31;
  const int wave = threadIdx.x >> 5;
  const int wr = wave >> 1, wc = wave & 1;
  const int tBase = tT * 128 + wr * 64;
  const int eBase = eT * 128 + wc * 64;
  const unsigned short* Pb = P + (size_t)b * T_ * T_;
  const unsigned short* Vb = Vt + (size_t)b * E_ * T_;

  v8f acc[4][4] = {};
  const int sEnd = (tT + 1) * 128;  // causal: P is zero beyond this
  for (int s0 = 0; s0 < sEnd; s0 += 32) {
    BFrag a[4], bb[4];
#pragma unroll
    for (int mr = 0; mr < 4; ++mr) load_fragA(Pb, tBase + mr * 16, T_, s0, lane, a[mr]);
#pragma unroll
    for (int nc = 0; nc < 4; ++nc) load_fragB(Vb, eBase + nc * 16, T_, s0, lane, bb[nc]);
#pragma unroll
    for (int mr = 0; mr < 4; ++mr)
#pragma unroll
      for (int nc = 0; nc < 4; ++nc)
        acc[mr][nc] = wmma_bf16(a[mr], bb[nc], acc[mr][nc]);
  }

  float invZ = 1.0f / Z[b];
#pragma unroll
  for (int mr = 0; mr < 4; ++mr)
#pragma unroll
    for (int nc = 0; nc < 4; ++nc)
#pragma unroll
      for (int r = 0; r < 8; ++r) {
        int t = tBase + mr * 16 + r + 8 * (lane >> 4);
        int e = eBase + nc * 16 + (lane & 15);
        out[((size_t)b * T_ + t) * E_ + e] = acc[mr][nc][r] * invZ;
      }
}

extern "C" void kernel_launch(void* const* d_in, const int* in_sizes, int n_in,
                              void* d_out, int out_size, void* d_ws, size_t ws_size,
                              hipStream_t stream) {
  const float* x  = (const float*)d_in[0];
  const float* Wq = (const float*)d_in[1];
  const float* bq = (const float*)d_in[2];
  const float* Wk = (const float*)d_in[3];
  const float* bk = (const float*)d_in[4];
  const float* Wv = (const float*)d_in[5];
  const float* bv = (const float*)d_in[6];
  float* out = (float*)d_out;

  char* ws = (char*)d_ws;
  const size_t MB = 1024ull * 1024ull;
  unsigned short* Qb = (unsigned short*)(ws);             // 32 MB [B*T][D] bf16 (pre-scaled)
  unsigned short* Kb = (unsigned short*)(ws + 32 * MB);   // 32 MB [B*T][D] bf16
  unsigned short* Vt = (unsigned short*)(ws + 64 * MB);   // 32 MB [B][E][T] bf16 (transposed)
  unsigned short* P  = (unsigned short*)(ws + 96 * MB);   // 64 MB [B][T][T] bf16
  unsigned short* Xb = (unsigned short*)(ws + 160 * MB);  // 32 MB [B*T][D] bf16
  unsigned short* Wt = (unsigned short*)(ws + 192 * MB);  //  6 MB 3x[E][D] bf16 (transposed)
  unsigned int*   mE = (unsigned int*)(ws + 200 * MB);    // per-batch encoded max
  float*          Z  = (float*)(ws + 200 * MB + 256);     // per-batch sum

  k_init<<<dim3(1), dim3(32), 0, stream>>>(mE, Z);

  int n4 = (B_ * T_ * D_) / 4;
  k_cvt_x<<<dim3((n4 + 255) / 256), dim3(256), 0, stream>>>(x, Xb, n4);
  k_cvt_wt<<<dim3((D_ * E_) / 256, 3), dim3(256), 0, stream>>>(Wq, Wk, Wv, Wt);

  k_qkv<<<dim3(E_ / 128, (B_ * T_) / 128, 3), dim3(128), 0, stream>>>(
      Xb, Wt, bq, bk, bv, Qb, Kb, Vt);

  k_smax<<<dim3(T_ / 128, T_ / 128, B_), dim3(128), 0, stream>>>(Qb, Kb, mE);
  k_pexp<<<dim3(T_ / 128, T_ / 128, B_), dim3(128), 0, stream>>>(Qb, Kb, mE, P, Z);
  k_pv  <<<dim3(E_ / 128, T_ / 128, B_), dim3(128), 0, stream>>>(P, Vt, Z, out);
}
